// DGCNN_15788299780189
// MI455X (gfx1250) — matrix operand order repaired and checked by reading, not compiled, not run on hardware
//
#include <hip/hip_runtime.h>
#include <cstdint>

#define LRELU_SLOPE 0.2f
#define BN_EPS 1e-5f

typedef __attribute__((ext_vector_type(16))) __bf16 v16bf;
typedef __attribute__((ext_vector_type(8)))  float  v8f;
typedef __attribute__((ext_vector_type(8)))  unsigned short us8;
typedef __attribute__((ext_vector_type(4)))  unsigned short us4;
typedef __attribute__((ext_vector_type(4)))  float  f4;

union BFrag { v16bf v; unsigned short s[16]; us8 h[2]; };

__device__ __forceinline__ unsigned short f2bf(float f) {
  unsigned u = __float_as_uint(f);
  u += 0x7fffu + ((u >> 16) & 1u);        // round-to-nearest-even
  return (unsigned short)(u >> 16);
}

__device__ __forceinline__ v8f wmma_bf16(const BFrag& a, const BFrag& b, v8f c) {
  // D(16x16 f32) = A(16x32 bf16) * B(32x16 bf16) + C
  return __builtin_amdgcn_wmma_f32_16x16x32_bf16(false, a.v, false, b.v,
                                                 (short)0, c, false, false);
}

// A fragment from LDS: 16x32 bf16 tile at K-chunk kc, row stride `stride` shorts.
// Lane l holds row m=l&15; element e -> K = (e/8)*16 + (l>>4)*8 + (e&7).
// => two 16-byte contiguous LDS reads per lane.
__device__ __forceinline__ void load_a_frag(const unsigned short* lds, int stride,
                                            int kc, int lane, BFrag& f) {
  const int m  = lane & 15;
  const int kb = (lane >> 4) << 3;
  const unsigned short* base = lds + m * stride + (kc << 5) + kb;
  f.h[0] = *(const us8*)(base);
  f.h[1] = *(const us8*)(base + 16);
}

// B fragment: 32x16 (K x Nout). Column n = weight row o (row-major O x Cpad bf16).
// Lane l holds column l&15; element e -> K = (e/8)*16 + (l>>4)*8 + (e&7) => two 16-byte reads.
__device__ __forceinline__ void load_b_frag(const unsigned short* __restrict__ W,
                                            int o, int Cpad, int kc, int lane, BFrag& f) {
  const unsigned short* base = W + (size_t)o * Cpad + (kc << 5) + ((lane >> 4) << 3);
  f.h[0] = *(const us8*)(base);
  f.h[1] = *(const us8*)(base + 16);
}

// ---------------- helper kernels ----------------

__global__ void xpose3_kernel(const float* __restrict__ x, float* __restrict__ xt,
                              int N, int total) {
  int gid = blockIdx.x * blockDim.x + threadIdx.x;
  if (gid >= total) return;
  int b = gid / N, n = gid - b * N;
#pragma unroll
  for (int c = 0; c < 3; ++c)
    xt[(size_t)gid * 3 + c] = x[((size_t)b * 3 + c) * N + n];
}

__global__ void cvtw_kernel(const float* __restrict__ src, unsigned short* __restrict__ dst,
                            int Cin, int Cpad, int total) {
  int gid = blockIdx.x * blockDim.x + threadIdx.x;
  if (gid >= total) return;
  int o = gid / Cpad, c = gid - o * Cpad;
  dst[gid] = (c < Cin) ? f2bf(src[(size_t)o * Cin + c]) : (unsigned short)0;
}

template <int C>
__global__ void sqnorm_kernel(const float* __restrict__ xin, int stride,
                              float* __restrict__ xx, int total) {
  int gid = blockIdx.x * blockDim.x + threadIdx.x;
  if (gid >= total) return;
  const float* r = xin + (size_t)gid * stride;
  float s = 0.f;
#pragma unroll 4
  for (int c = 0; c < C; ++c) s += r[c] * r[c];
  xx[gid] = s;
}

// One thread per point: streaming top-20 of dist = 2*inner - ||xi||^2 - ||xj||^2.
template <int C>
__global__ void knn_kernel(const float* __restrict__ xin, int stride,
                           const float* __restrict__ xx, int* __restrict__ idx,
                           int N, int total) {
  int gid = blockIdx.x * blockDim.x + threadIdx.x;
  if (gid >= total) return;
  int b = gid / N, n = gid - b * N;
  const float* xb = xin + (size_t)b * N * stride;
  const float* rowp = xb + (size_t)n * stride;
  float ctrv[C];
#pragma unroll 4
  for (int c = 0; c < C; ++c) ctrv[c] = rowp[c];
  const float* xxb = xx + (size_t)b * N;
  float xxn = xxb[n];
  float bestd[20]; int besti[20];
#pragma unroll
  for (int t = 0; t < 20; ++t) { bestd[t] = -3.4e38f; besti[t] = 0; }
  for (int m = 0; m < N; ++m) {
    const float* mp = xb + (size_t)m * stride;
    float inner = 0.f;
#pragma unroll 4
    for (int c = 0; c < C; ++c) inner += ctrv[c] * mp[c];
    float d = 2.f * inner - xxn - xxb[m];
    if (d > bestd[19]) {                   // strict > => earlier index wins ties
      int pos = 19;
      while (pos > 0 && bestd[pos - 1] < d) {
        bestd[pos] = bestd[pos - 1]; besti[pos] = besti[pos - 1]; --pos;
      }
      bestd[pos] = d; besti[pos] = m;
    }
  }
  int* op = idx + (size_t)gid * 20;
#pragma unroll
  for (int t = 0; t < 20; ++t) op[t] = besti[t];
}

// ---------------- fused edge conv: gather -> WMMA GEMM -> BN/LReLU -> max_k ----
// Workgroup = one batch x 16 points, 4 waves. Wave w owns output chunks
// {w, w+4, ...} of 16 channels each (CPW = O/64 chunks per wave).
template <int C, int Cpad, int O>
__global__ __launch_bounds__(128) void edge_conv_kernel(
    const float* __restrict__ xin, int in_stride,
    const int* __restrict__ idx,
    const unsigned short* __restrict__ Wbf,
    const float* __restrict__ gamma, const float* __restrict__ beta,
    float* __restrict__ xout, int out_stride, int N) {
  constexpr int CPW = O / 64;              // 16-col output chunks per wave
  constexpr int Kc  = Cpad / 32;           // K chunks
  constexpr bool HOIST = (CPW * Kc) <= 8;  // keep B fragments in regs across k
  constexpr int  BW = HOIST ? CPW * Kc : 1;
  constexpr int C2 = 2 * C;

  __shared__ __align__(16) unsigned short Alds[16 * Cpad];
  __shared__ float Ctr[16 * C];
  __shared__ int   Jrow[16];

  const int b = blockIdx.y;
  const int n0 = blockIdx.x << 4;
  const int tid = threadIdx.x;
  const int w = tid >> 5, lane = tid & 31;

  // stage the 16 center rows once
  for (int t = tid; t < 16 * C; t += 128) {
    int r = t / C, c = t - r * C;
    Ctr[t] = xin[((size_t)b * N + n0 + r) * in_stride + c];
  }

  float sc[CPW], bi[CPW];
  const float bninv = rsqrtf(1.f + BN_EPS);
#pragma unroll
  for (int i = 0; i < CPW; ++i) {
    int o = ((w + 4 * i) << 4) + (lane & 15);
    sc[i] = gamma[o] * bninv;
    bi[i] = beta[o];
  }

  BFrag bw[BW];
  if (HOIST) {
#pragma unroll
    for (int i = 0; i < CPW; ++i) {
      int o = ((w + 4 * i) << 4) + (lane & 15);
#pragma unroll
      for (int kc = 0; kc < Kc; ++kc)
        load_b_frag(Wbf, o, Cpad, kc, lane, bw[i * Kc + kc]);
    }
  }

  v8f macc[CPW];
#pragma unroll
  for (int i = 0; i < CPW; ++i)
#pragma unroll
    for (int e = 0; e < 8; ++e) macc[i][e] = -3.4e38f;

  for (int k = 0; k < 20; ++k) {
    if (tid < 16) Jrow[tid] = idx[((size_t)b * N + n0 + tid) * 20 + k];
    __syncthreads();   // Jrow ready; everyone done reading previous Alds
    // build edge-feature tile [nbr - ctr | ctr | 0-pad] as bf16 in LDS
    for (int t = tid; t < 16 * Cpad; t += 128) {
      int r = t / Cpad, c = t - r * Cpad;
      float v = 0.f;
      if (c < C) {
        v = xin[((size_t)b * N + Jrow[r]) * in_stride + c] - Ctr[r * C + c];
      } else if (c < C2) {
        v = Ctr[r * C + (c - C)];
      }
      Alds[t] = f2bf(v);
    }
    __syncthreads();

    v8f acc0[CPW], acc1[CPW];
#pragma unroll
    for (int i = 0; i < CPW; ++i)
#pragma unroll
      for (int e = 0; e < 8; ++e) { acc0[i][e] = 0.f; acc1[i][e] = 0.f; }

#pragma unroll
    for (int kc = 0; kc < Kc; ++kc) {
      BFrag a; load_a_frag(Alds, Cpad, kc, lane, a);
#pragma unroll
      for (int i = 0; i < CPW; ++i) {
        BFrag bf;
        if (HOIST) {
          bf = bw[i * Kc + kc];
        } else {
          int o = ((w + 4 * i) << 4) + (lane & 15);
          load_b_frag(Wbf, o, Cpad, kc, lane, bf);
        }
        if (kc & 1) acc1[i] = wmma_bf16(a, bf, acc1[i]);
        else        acc0[i] = wmma_bf16(a, bf, acc0[i]);
      }
    }
#pragma unroll
    for (int i = 0; i < CPW; ++i) {
#pragma unroll
      for (int e = 0; e < 8; ++e) {
        float y = (acc0[i][e] + acc1[i][e]) * sc[i] + bi[i];
        y = (y >= 0.f) ? y : LRELU_SLOPE * y;
        macc[i][e] = fmaxf(macc[i][e], y);
      }
    }
  }

  const int rbase = (lane >> 4) << 3;
#pragma unroll
  for (int i = 0; i < CPW; ++i) {
    int o = ((w + 4 * i) << 4) + (lane & 15);
#pragma unroll
    for (int e = 0; e < 8; ++e)
      xout[((size_t)b * N + n0 + rbase + e) * out_stride + o] = macc[i][e];
  }
}

// ---------------- W5 (512->1024) + BN/LReLU + fused global max/mean pool ------
// Workgroup = one batch x 64 output channels (4 waves x 16 cols); loops n-tiles.
// All 16 weight B-fragments live in registers for the whole loop.
__global__ __launch_bounds__(128) void pool_conv_kernel(
    const float* __restrict__ xcat, const unsigned short* __restrict__ W5bf,
    const float* __restrict__ gamma, const float* __restrict__ beta,
    float* __restrict__ p, int N) {
  __shared__ __align__(16) unsigned short Alds[16 * 512];
  const int b = blockIdx.y;
  const int tid = threadIdx.x, w = tid >> 5, lane = tid & 31;
  const int o = (blockIdx.x << 6) + (w << 4) + (lane & 15);
  const float sc = gamma[o] * rsqrtf(1.f + BN_EPS);
  const float bi = beta[o];

  BFrag bw[16];
#pragma unroll
  for (int kc = 0; kc < 16; ++kc) load_b_frag(W5bf, o, 512, kc, lane, bw[kc]);

  float vmax = -3.4e38f, vsum = 0.f;
  for (int n0 = 0; n0 < N; n0 += 16) {
    // stage 16x512 f32 tile -> bf16 LDS (float4 loads, 8-byte LDS stores)
    for (int t = tid; t < 16 * 128; t += 128) {
      int r = t >> 7, c4 = t & 127;
      f4 v = ((const f4*)(xcat + ((size_t)b * N + n0 + r) * 512))[c4];
      us4 hh;
#pragma unroll
      for (int j = 0; j < 4; ++j) hh[j] = f2bf(v[j]);
      ((us4*)Alds)[t] = hh;
    }
    __syncthreads();
    if (n0 + 16 < N)
      __builtin_prefetch(&xcat[((size_t)b * N + n0 + 16) * 512], 0, 0);

    v8f acc[4];
#pragma unroll
    for (int q = 0; q < 4; ++q)
#pragma unroll
      for (int e = 0; e < 8; ++e) acc[q][e] = 0.f;
#pragma unroll
    for (int kc = 0; kc < 16; ++kc) {
      BFrag a; load_a_frag(Alds, 512, kc, lane, a);
      acc[kc & 3] = wmma_bf16(a, bw[kc], acc[kc & 3]);
    }
#pragma unroll
    for (int e = 0; e < 8; ++e) {
      float y = (acc[0][e] + acc[1][e] + acc[2][e] + acc[3][e]) * sc + bi;
      y = (y >= 0.f) ? y : LRELU_SLOPE * y;
      vmax = fmaxf(vmax, y);
      vsum += y;
    }
    __syncthreads();
  }
  // lanes l and l^16 hold rows 0-7 / 8-15 of the same column: merge halves
  float omax = fmaxf(vmax, __shfl_xor(vmax, 16));
  float osum = vsum + __shfl_xor(vsum, 16);
  if (lane < 16) {
    p[(size_t)b * 2048 + o] = omax;
    p[(size_t)b * 2048 + 1024 + o] = osum / (float)N;
  }
}

// ---------------- small FC head ----------------
__global__ void fc_kernel(const float* __restrict__ in, const float* __restrict__ W,
                          const float* __restrict__ bias, const float* __restrict__ gamma,
                          const float* __restrict__ beta, float* __restrict__ out,
                          int Bn, int IN, int OUT, int do_lrelu) {
  int gid = blockIdx.x * blockDim.x + threadIdx.x;
  if (gid >= Bn * OUT) return;
  int b = gid / OUT, o = gid - b * OUT;
  const float* ir = in + (size_t)b * IN;
  const float* wr = W + (size_t)o * IN;
  float s = 0.f;
#pragma unroll 4
  for (int c = 0; c < IN; ++c) s += ir[c] * wr[c];
  if (bias) s += bias[o];
  if (gamma) s = s * (gamma[o] * rsqrtf(1.f + BN_EPS)) + beta[o];
  if (do_lrelu) s = (s >= 0.f) ? s : LRELU_SLOPE * s;
  out[gid] = s;
}

// ---------------- launcher ----------------
extern "C" void kernel_launch(void* const* d_in, const int* in_sizes, int n_in,
                              void* d_out, int out_size, void* d_ws, size_t ws_size,
                              hipStream_t stream) {
  (void)in_sizes; (void)n_in; (void)out_size; (void)ws_size;
  const int B = 16, N = 2048;
  const float* x   = (const float*)d_in[0];
  const float* W1  = (const float*)d_in[1];
  const float* W2  = (const float*)d_in[2];
  const float* W3  = (const float*)d_in[3];
  const float* W4  = (const float*)d_in[4];
  const float* W5  = (const float*)d_in[5];
  const float* g1  = (const float*)d_in[6];  const float* b1 = (const float*)d_in[7];
  const float* g2  = (const float*)d_in[8];  const float* b2 = (const float*)d_in[9];
  const float* g3  = (const float*)d_in[10]; const float* b3 = (const float*)d_in[11];
  const float* g4  = (const float*)d_in[12]; const float* b4 = (const float*)d_in[13];
  const float* g5  = (const float*)d_in[14]; const float* b5 = (const float*)d_in[15];
  const float* g6  = (const float*)d_in[16]; const float* b6 = (const float*)d_in[17];
  const float* g7  = (const float*)d_in[18]; const float* b7 = (const float*)d_in[19];
  const float* L1w = (const float*)d_in[20];
  const float* L2w = (const float*)d_in[21]; const float* L2b = (const float*)d_in[22];
  const float* L3w = (const float*)d_in[23]; const float* L3b = (const float*)d_in[24];

  char* wsb = (char*)d_ws;
  size_t off = 0;
  auto carve = [&](size_t bytes) -> void* {
    void* pp = wsb + off;
    off += (bytes + 255) & ~(size_t)255;
    return pp;
  };
  float* xt0  = (float*)carve((size_t)B * N * 3 * 4);
  float* xcat = (float*)carve((size_t)B * N * 512 * 4);  // x1|x2|x3|x4 column slices
  float* xx   = (float*)carve((size_t)B * N * 4);
  int*   nidx = (int*)carve((size_t)B * N * 20 * 4);
  float* pbuf = (float*)carve((size_t)B * 2048 * 4);
  float* h1   = (float*)carve((size_t)B * 512 * 4);
  float* h2   = (float*)carve((size_t)B * 256 * 4);
  unsigned short* w1b = (unsigned short*)carve((size_t)64 * 32 * 2);
  unsigned short* w2b = (unsigned short*)carve((size_t)64 * 128 * 2);
  unsigned short* w3b = (unsigned short*)carve((size_t)128 * 128 * 2);
  unsigned short* w4b = (unsigned short*)carve((size_t)256 * 256 * 2);
  unsigned short* w5b = (unsigned short*)carve((size_t)1024 * 512 * 2);

  const int BN = B * N;
  xpose3_kernel<<<BN / 256, 256, 0, stream>>>(x, xt0, N, BN);
  cvtw_kernel<<<(64 * 32) / 256, 256, 0, stream>>>(W1, w1b, 6, 32, 64 * 32);
  cvtw_kernel<<<(64 * 128) / 256, 256, 0, stream>>>(W2, w2b, 128, 128, 64 * 128);
  cvtw_kernel<<<(128 * 128) / 256, 256, 0, stream>>>(W3, w3b, 128, 128, 128 * 128);
  cvtw_kernel<<<(256 * 256) / 256, 256, 0, stream>>>(W4, w4b, 256, 256, 256 * 256);
  cvtw_kernel<<<(1024 * 512) / 256, 256, 0, stream>>>(W5, w5b, 512, 512, 1024 * 512);

  dim3 eg(N / 16, B);
  // stage 1: 3 -> 64 (cols 0..63)
  sqnorm_kernel<3><<<BN / 256, 256, 0, stream>>>(xt0, 3, xx, BN);
  knn_kernel<3><<<BN / 256, 256, 0, stream>>>(xt0, 3, xx, nidx, N, BN);
  edge_conv_kernel<3, 32, 64><<<eg, 128, 0, stream>>>(xt0, 3, nidx, w1b, g1, b1,
                                                      xcat, 512, N);
  // stage 2: 64 -> 64 (cols 64..127)
  sqnorm_kernel<64><<<BN / 256, 256, 0, stream>>>(xcat, 512, xx, BN);
  knn_kernel<64><<<BN / 256, 256, 0, stream>>>(xcat, 512, xx, nidx, N, BN);
  edge_conv_kernel<64, 128, 64><<<eg, 128, 0, stream>>>(xcat, 512, nidx, w2b, g2, b2,
                                                        xcat + 64, 512, N);
  // stage 3: 64 -> 128 (cols 128..255)
  sqnorm_kernel<64><<<BN / 256, 256, 0, stream>>>(xcat + 64, 512, xx, BN);
  knn_kernel<64><<<BN / 256, 256, 0, stream>>>(xcat + 64, 512, xx, nidx, N, BN);
  edge_conv_kernel<64, 128, 128><<<eg, 128, 0, stream>>>(xcat + 64, 512, nidx, w3b, g3, b3,
                                                         xcat + 128, 512, N);
  // stage 4: 128 -> 256 (cols 256..511)
  sqnorm_kernel<128><<<BN / 256, 256, 0, stream>>>(xcat + 128, 512, xx, BN);
  knn_kernel<128><<<BN / 256, 256, 0, stream>>>(xcat + 128, 512, xx, nidx, N, BN);
  edge_conv_kernel<128, 256, 256><<<eg, 128, 0, stream>>>(xcat + 128, 512, nidx, w4b, g4, b4,
                                                          xcat + 256, 512, N);
  // W5 + BN/LReLU + fused max/mean pooling -> p (B,2048)
  pool_conv_kernel<<<dim3(16, B), 128, 0, stream>>>(xcat, w5b, g5, b5, pbuf, N);
  // FC head
  fc_kernel<<<(B * 512 + 255) / 256, 256, 0, stream>>>(pbuf, L1w, nullptr, g6, b6, h1,
                                                       B, 2048, 512, 1);
  fc_kernel<<<(B * 256 + 255) / 256, 256, 0, stream>>>(h1, L2w, L2b, g7, b7, h2,
                                                       B, 512, 256, 1);
  fc_kernel<<<(B * 40 + 255) / 256, 256, 0, stream>>>(h2, L3w, L3b, nullptr, nullptr,
                                                      (float*)d_out, B, 256, 40, 0);
}
